// rnn_30150670417996
// MI455X (gfx1250) — matrix operand order repaired and checked by reading, not compiled, not run on hardware
//
#include <hip/hip_runtime.h>

// Elman RNN scan: h_{t+1} = tanh( wih*h_t + whh*h_t terms ), see reference.
// S=512, B=2048, D=64, H=128.  Persistent time loop per wave; each wave owns a
// 16-batch-row WMMA tile for all 512 steps (independent recurrence chains).

#define S_LEN 512
#define BATCH 2048
#define DIN   64
#define HID   128
#define WAVES 2            // waves per block; each wave = one 16-row batch tile
#define TPB   (WAVES * 32) // wave32

typedef __attribute__((ext_vector_type(16))) __bf16 v16bf;
typedef __attribute__((ext_vector_type(8)))  __bf16 v8bf;
typedef __attribute__((ext_vector_type(8)))  float  v8f;

__device__ __forceinline__ v8f splat8(float s) {
    v8f r;
#pragma unroll
    for (int i = 0; i < 8; ++i) r[i] = s;
    return r;
}

// tanh(x) = 1 - 2/(exp(2x)+1): one v_exp_f32 (trans) + v_rcp_f32 + FMAs
__device__ __forceinline__ float fast_tanhf(float x) {
    float e = __expf(2.0f * x);
    return 1.0f - 2.0f * __builtin_amdgcn_rcpf(e + 1.0f);
}

__global__ __launch_bounds__(TPB)
void elman_scan_kernel(const float* __restrict__ x,    // [S,B,D]
                       const float* __restrict__ h0,   // [B,H]
                       const float* __restrict__ Wih,  // [H,D] row-major
                       const float* __restrict__ bih,  // [H]
                       const float* __restrict__ Whh,  // [H,H] row-major
                       const float* __restrict__ bhh,  // [H]
                       float* __restrict__ hout)       // [B,H]
{
    // bf16 weight images in LDS (loaded once; ~57 KB total, one WGP block)
    __shared__ __bf16 sWih[HID * DIN];        // 16 KB, row-major [n][k]
    __shared__ __bf16 sWhh[HID * HID];        // 32 KB, row-major [n][k]
    __shared__ float  sBih[HID];
    __shared__ float  sBhh[HID];
    __shared__ __bf16 sWX[WAVES][16 * HID];   // per-wave wih_x staging (C->A transpose)

    const int tid  = threadIdx.x;
    const int wave = tid >> 5;
    const int lane = tid & 31;
    const int half = lane >> 4;   // 0: lanes 0-15, 1: lanes 16-31
    const int l15  = lane & 15;
    const int b0   = (blockIdx.x * WAVES + wave) * 16;  // batch tile base

    for (int i = tid; i < HID * DIN; i += TPB) sWih[i] = (__bf16)Wih[i];
    for (int i = tid; i < HID * HID; i += TPB) sWhh[i] = (__bf16)Whh[i];
    for (int i = tid; i < HID;       i += TPB) { sBih[i] = bih[i]; sBhh[i] = bhh[i]; }
    __syncthreads();

    // h in WMMA C/D layout: 8 n-tiles; elem j of lane -> row j+8*half, col l15+16*nt
    v8f hacc[8];
#pragma unroll
    for (int nt = 0; nt < 8; ++nt)
#pragma unroll
        for (int j = 0; j < 8; ++j)
            hacc[nt][j] = h0[(size_t)(b0 + j + 8 * half) * HID + nt * 16 + l15];

    __bf16* wx = &sWX[wave][0];

    for (int t = 0; t < S_LEN; ++t) {
        const float* xr = x + ((size_t)t * BATCH + b0 + l15) * DIN;

        // prefetch next timestep's x rows (global_prefetch_b8)
        if (t + 1 < S_LEN) {
            const float* xn = x + ((size_t)(t + 1) * BATCH + b0 + l15) * DIN;
            __builtin_prefetch(xn, 0, 1);
            __builtin_prefetch(xn + 32, 0, 1);
        }

        // ---- A fragments of x_t (16x64 bf16 -> two K=32 fragments) ----
        // A layout: lane l15 row, elems 0..15 = K 16h..16h+15 (contiguous)
        v16bf xa[2];
#pragma unroll
        for (int f = 0; f < 2; ++f) {
            const int c0 = f * 32 + half * 16;
            const float4 u0 = *(const float4*)(xr + c0);
            const float4 u1 = *(const float4*)(xr + c0 + 4);
            const float4 u2 = *(const float4*)(xr + c0 + 8);
            const float4 u3 = *(const float4*)(xr + c0 + 12);
            v16bf a;
            a[0]=(__bf16)u0.x; a[1]=(__bf16)u0.y; a[2]=(__bf16)u0.z; a[3]=(__bf16)u0.w;
            a[4]=(__bf16)u1.x; a[5]=(__bf16)u1.y; a[6]=(__bf16)u1.z; a[7]=(__bf16)u1.w;
            a[8]=(__bf16)u2.x; a[9]=(__bf16)u2.y; a[10]=(__bf16)u2.z; a[11]=(__bf16)u2.w;
            a[12]=(__bf16)u3.x; a[13]=(__bf16)u3.y; a[14]=(__bf16)u3.z; a[15]=(__bf16)u3.w;
            xa[f] = a;
        }

        // ---- GEMM1: wih_x = (x_t @ W_ih^T + b_ih) * h ----
        // B fragment (B = W_ih^T): lane -> col n0+l15, K 32kt+16h.. = contiguous 32B in LDS
        v8f wihx[8];
#pragma unroll
        for (int nt = 0; nt < 8; ++nt) {
            v8f c = splat8(sBih[nt * 16 + l15]);
#pragma unroll
            for (int kt = 0; kt < 2; ++kt) {
                const v16bf bfrag =
                    *(const v16bf*)(sWih + (nt * 16 + l15) * DIN + kt * 32 + half * 16);
                c = __builtin_amdgcn_wmma_f32_16x16x32_bf16(
                        false, xa[kt], false, bfrag, (short)0, c, false, false);
            }
            v8f wxv = c * hacc[nt];       // elementwise * h (layouts match)
            wihx[nt] = wxv;               // keep f32 copy for the tanh sum
            // stage bf16 row-major [m][k] into LDS for GEMM2 A fragments
#pragma unroll
            for (int j = 0; j < 8; ++j)
                wx[(j + 8 * half) * HID + nt * 16 + l15] = (__bf16)wxv[j];
        }

        // wave-private RAW through LDS: DS is in-order per wave; fence the compiler
        asm volatile("s_wait_dscnt 0" ::: "memory");

        // ---- A fragments of wih_x (16x128 -> four K=32 fragments) ----
        v16bf wa[4];
#pragma unroll
        for (int f = 0; f < 4; ++f) {
            const __bf16* p = wx + l15 * HID + f * 32 + half * 16;
            union { v16bf v; v8bf h[2]; } fr;
            fr.h[0] = *(const v8bf*)(p);
            fr.h[1] = *(const v8bf*)(p + 8);
            wa[f] = fr.v;
        }

        // ---- GEMM2: whh = (wih_x @ W_hh^T + b_hh) * h ; h = tanh(wih_x + whh) ----
#pragma unroll
        for (int nt = 0; nt < 8; ++nt) {
            v8f c = splat8(sBhh[nt * 16 + l15]);
#pragma unroll
            for (int kt = 0; kt < 4; ++kt) {
                const v16bf bfrag =
                    *(const v16bf*)(sWhh + (nt * 16 + l15) * HID + kt * 32 + half * 16);
                c = __builtin_amdgcn_wmma_f32_16x16x32_bf16(
                        false, wa[kt], false, bfrag, (short)0, c, false, false);
            }
            v8f s = wihx[nt] + c * hacc[nt];
#pragma unroll
            for (int j = 0; j < 8; ++j) hacc[nt][j] = fast_tanhf(s[j]);
        }

        // keep next iteration's LDS stores behind this iteration's fragment reads
        asm volatile("s_wait_dscnt 0" ::: "memory");
    }

    // ---- final hidden state, f32 [B,H] ----
#pragma unroll
    for (int nt = 0; nt < 8; ++nt)
#pragma unroll
        for (int j = 0; j < 8; ++j)
            hout[(size_t)(b0 + j + 8 * half) * HID + nt * 16 + l15] = hacc[nt][j];
}

extern "C" void kernel_launch(void* const* d_in, const int* in_sizes, int n_in,
                              void* d_out, int out_size, void* d_ws, size_t ws_size,
                              hipStream_t stream) {
    const float* x   = (const float*)d_in[0];
    const float* h0  = (const float*)d_in[1];
    const float* Wih = (const float*)d_in[2];
    const float* bih = (const float*)d_in[3];
    const float* Whh = (const float*)d_in[4];
    const float* bhh = (const float*)d_in[5];
    float* out = (float*)d_out;

    dim3 grid(BATCH / (16 * WAVES));   // 64 blocks x 2 waves = 128 independent chains
    dim3 block(TPB);
    elman_scan_kernel<<<grid, block, 0, stream>>>(x, h0, Wih, bih, Whh, bhh, out);
}
